// Tconv_block_57690000720235
// MI455X (gfx1250) — matrix operand order repaired and checked by reading, not compiled, not run on hardware
//
#include <hip/hip_runtime.h>
#include <hip/hip_bf16.h>

typedef _Float16 v16h __attribute__((ext_vector_type(16)));
typedef _Float16 v8h  __attribute__((ext_vector_type(8)));
typedef float    v8f  __attribute__((ext_vector_type(8)));

#define CIN   256
#define COUT  256
#define CMID  64
#define Hh    64
#define Ww    64
#define HO    128
#define WO    128
#define NB    4

// LDS pitches (halves): channel-contiguous, pitch chosen so pixel stride is
// 132 dwords (== 4 mod 64) -> conflict-free ds_load_b128 across 16 lanes.
#define XPITCH 264   // 256 + 8
#define MPITCH 72    // 64 + 8

// ws layout (bytes)
#define WS_MID   0
#define WS_WH    (2*1024*1024)                 // 9*16*8*512 halves = 1,179,648 B
#define WS_WL2   (WS_WH + 1179648)             // 9*16*2*512 halves =   294,912 B
#define WS_WL1   (WS_WL2 + 294912)             // 4*8*512   halves =    32,768 B

static __device__ __forceinline__ v16h lds_bfrag(const _Float16* p) {
    v8h lo = *(const v8h*)p;
    v8h hi = *(const v8h*)(p + 16);
    return __builtin_shufflevector(lo, hi, 0,1,2,3,4,5,6,7,8,9,10,11,12,13,14,15);
}

static __device__ __forceinline__ v8f wmma_f16(v16h a, v16h b, v8f c) {
    return __builtin_amdgcn_wmma_f32_16x16x32_f16(false, a, false, b, (short)0, c, false, false);
}

// A-matrix pack: lane l (M = l&15) holds 16 halves; element e -> K = (e<8?e:e+8) + 8*(l>>4)
__global__ void pack_w_high(const float* __restrict__ w, _Float16* __restrict__ dst) {
    int idx = blockIdx.x * 256 + threadIdx.x;          // < 9*16*8*512
    int e       =  idx        & 15;
    int lane    = (idx >> 4)  & 31;
    int cinblk  = (idx >> 9)  & 7;
    int coutblk = (idx >> 12) & 15;
    int tap     =  idx >> 16;                          // 0..8
    int K  = (e < 8 ? e : e + 8) + ((lane >> 4) << 3);
    int co = coutblk * 16 + (lane & 15);
    int ci = cinblk * 32 + K;
    int ky = tap / 3, kx = tap % 3;
    dst[idx] = (_Float16)w[((ci * COUT + co) * 3 + ky) * 3 + kx];
}

__global__ void pack_w_low2(const float* __restrict__ w, _Float16* __restrict__ dst) {
    int idx = blockIdx.x * 256 + threadIdx.x;          // < 9*16*2*512
    int e       =  idx        & 15;
    int lane    = (idx >> 4)  & 31;
    int cinblk  = (idx >> 9)  & 1;
    int coutblk = (idx >> 10) & 15;
    int tap     =  idx >> 14;                          // 0..8
    int K  = (e < 8 ? e : e + 8) + ((lane >> 4) << 3);
    int co = coutblk * 16 + (lane & 15);
    int ci = cinblk * 32 + K;                          // ci < 64
    int ky = tap / 3, kx = tap % 3;
    dst[idx] = (_Float16)w[((ci * COUT + co) * 3 + ky) * 3 + kx];
}

__global__ void pack_w_low1(const float* __restrict__ w, _Float16* __restrict__ dst) {
    int idx = blockIdx.x * 256 + threadIdx.x;          // < 4*8*512
    int e      =  idx        & 15;
    int lane   = (idx >> 4)  & 31;
    int cinblk = (idx >> 9)  & 7;
    int cmblk  =  idx >> 12;                           // 0..3
    int K  = (e < 8 ? e : e + 8) + ((lane >> 4) << 3);
    int cm = cmblk * 16 + (lane & 15);
    int ci = cinblk * 32 + K;
    dst[idx] = (_Float16)w[cm * CIN + ci];
}

// mid = w_low1 @ x + b_low1, stored as f16 [B][CMID][H*W]
__global__ __launch_bounds__(256) void mid_kernel(
        const float* __restrict__ x, const float* __restrict__ b_low1,
        const _Float16* __restrict__ wL1p, _Float16* __restrict__ mid) {
    __shared__ __align__(32) _Float16 xs[64 * XPITCH];
    int tid = threadIdx.x;
    int pixbase = blockIdx.x * 64;                     // flat pixel p = b*4096 + hw
    int b  = pixbase >> 12;
    int hw0 = pixbase & 4095;
    for (int it = 0; it < 64; ++it) {
        int idx = it * 256 + tid;
        int ci = idx >> 6, pix = idx & 63;
        xs[pix * XPITCH + ci] = (_Float16)x[(b * CIN + ci) * 4096 + hw0 + pix];
    }
    __syncthreads();
    int wv = tid >> 5, lane = tid & 31;
    int n = lane & 15, khi = (lane >> 4) << 3, rhi = (lane >> 4) << 3;
    for (int t = wv * 2; t < wv * 2 + 2; ++t) {
        int cmblk = t >> 2, pt = t & 3;
        v8f acc = {};
        #pragma unroll
        for (int cb = 0; cb < 8; ++cb) {
            v16h A = *(const v16h*)(wL1p + (cmblk * 8 + cb) * 512 + lane * 16);
            v16h Bm = lds_bfrag(xs + (pt * 16 + n) * XPITCH + cb * 32 + khi);
            acc = wmma_f16(A, Bm, acc);
        }
        int hw = hw0 + pt * 16 + n;
        #pragma unroll
        for (int r = 0; r < 8; ++r) {
            int cm = cmblk * 16 + r + rhi;
            mid[(b * CMID + cm) * 4096 + hw] = (_Float16)(acc[r] + b_low1[cm]);
        }
    }
}

// Fused stride-2 transposed conv (high + low), masked combine.
__global__ __launch_bounds__(256) void tconv_kernel(
        const float* __restrict__ x, const float* __restrict__ mask,
        const float* __restrict__ inv_mask,
        const float* __restrict__ b_high, const float* __restrict__ b_low2,
        const _Float16* __restrict__ wHp, const _Float16* __restrict__ wL2p,
        const _Float16* __restrict__ midp, float* __restrict__ out) {
    __shared__ __align__(32) _Float16 xs[2 * 17 * XPITCH];
    __shared__ __align__(32) _Float16 ms[2 * 17 * MPITCH];
    int tid = threadIdx.x;
    int n0 = (blockIdx.x & 3) * 16;
    int m  = (blockIdx.x >> 2) & 63;
    int b  =  blockIdx.x >> 8;

    // stage x rows {m, m+1}, cols n0..n0+16 (zero-fill OOB), f32 -> f16
    for (int it = 0; it < 34; ++it) {
        int idx = it * 256 + tid;                      // < 8704 = 256ch * 34
        int ci = idx / 34;
        int rem = idx - ci * 34;
        int row = rem >= 17 ? 1 : 0;
        int ix = rem - row * 17;
        int iy = m + row, gx = n0 + ix;
        float v = (iy < Hh && gx < Ww) ? x[((b * CIN + ci) * Hh + iy) * Ww + gx] : 0.0f;
        xs[(row * 17 + ix) * XPITCH + ci] = (_Float16)v;
    }
    // stage mid (f16) same footprint, 64 channels
    for (int it = 0; it < 9; ++it) {
        int idx = it * 256 + tid;
        if (idx < 2176) {                              // 64ch * 34
            int cm = idx / 34;
            int rem = idx - cm * 34;
            int row = rem >= 17 ? 1 : 0;
            int ix = rem - row * 17;
            int iy = m + row, gx = n0 + ix;
            _Float16 v = (iy < Hh && gx < Ww)
                ? midp[((b * CMID + cm) * Hh + iy) * Ww + gx] : (_Float16)0.0f;
            ms[(row * 17 + ix) * MPITCH + cm] = v;
        }
    }
    __syncthreads();

    int wv = tid >> 5, lane = tid & 31;
    int n = lane & 15, khi = (lane >> 4) << 3, rhi = (lane >> 4) << 3;

    for (int cbi = 0; cbi < 2; ++cbi) {
        int cb = wv * 2 + cbi;                         // cout block 0..15
        v8f accH[4] = {v8f{}, v8f{}, v8f{}, v8f{}};
        v8f accL[4] = {v8f{}, v8f{}, v8f{}, v8f{}};

        #pragma unroll
        for (int ky = 0; ky < 3; ++ky) {
            const int row_l = (ky == 0) ? 1 : 0;       // iy = m+1 only for ky==0
            const int py    = (ky == 1) ? 0 : 1;
            #pragma unroll
            for (int kx = 0; kx < 3; ++kx) {
                const int dx   = (kx == 0) ? 1 : 0;    // ix = n+1 only for kx==0
                const int px   = (kx == 1) ? 0 : 1;
                const int pidx = py * 2 + px;
                const int tap  = ky * 3 + kx;
                const int pixL = n + dx;
                const int xoff = (row_l * 17 + pixL) * XPITCH + khi;
                const int moff = (row_l * 17 + pixL) * MPITCH + khi;
                #pragma unroll
                for (int cblk = 0; cblk < 8; ++cblk) { // high: K = 256
                    v16h A = *(const v16h*)(wHp + ((tap * 16 + cb) * 8 + cblk) * 512 + lane * 16);
                    v16h Bm = lds_bfrag(xs + xoff + cblk * 32);
                    accH[pidx] = wmma_f16(A, Bm, accH[pidx]);
                }
                #pragma unroll
                for (int cblk = 0; cblk < 2; ++cblk) { // low: K = 64
                    v16h A = *(const v16h*)(wL2p + ((tap * 16 + cb) * 2 + cblk) * 512 + lane * 16);
                    v16h Bm = lds_bfrag(ms + moff + cblk * 32);
                    accL[pidx] = wmma_f16(A, Bm, accL[pidx]);
                }
            }
        }

        float bh[8], bl[8];
        #pragma unroll
        for (int r = 0; r < 8; ++r) {
            int co = cb * 16 + r + rhi;
            bh[r] = b_high[co];
            bl[r] = b_low2[co];
        }
        #pragma unroll
        for (int py = 0; py < 2; ++py) {
            #pragma unroll
            for (int px = 0; px < 2; ++px) {
                int pidx = py * 2 + px;
                int oy = 2 * m + py;
                int ox = 2 * (n0 + n) + px;
                float mk = mask[(b * HO + oy) * WO + ox];
                float im = inv_mask[(b * HO + oy) * WO + ox];
                #pragma unroll
                for (int r = 0; r < 8; ++r) {
                    int co = cb * 16 + r + rhi;
                    out[((b * COUT + co) * HO + oy) * WO + ox] =
                        mk * (accH[pidx][r] + bh[r]) + im * (accL[pidx][r] + bl[r]);
                }
            }
        }
    }
}

extern "C" void kernel_launch(void* const* d_in, const int* in_sizes, int n_in,
                              void* d_out, int out_size, void* d_ws, size_t ws_size,
                              hipStream_t stream) {
    const float* inx      = (const float*)d_in[0];
    const float* mask     = (const float*)d_in[1];
    const float* inv_mask = (const float*)d_in[2];
    const float* w_high   = (const float*)d_in[3];
    const float* b_high   = (const float*)d_in[4];
    const float* w_low1   = (const float*)d_in[5];
    const float* b_low1   = (const float*)d_in[6];
    const float* w_low2   = (const float*)d_in[7];
    const float* b_low2   = (const float*)d_in[8];
    float* out = (float*)d_out;

    char* wsb = (char*)d_ws;
    _Float16* midp = (_Float16*)(wsb + WS_MID);
    _Float16* wHp  = (_Float16*)(wsb + WS_WH);
    _Float16* wL2p = (_Float16*)(wsb + WS_WL2);
    _Float16* wL1p = (_Float16*)(wsb + WS_WL1);

    pack_w_high<<<2304, 256, 0, stream>>>(w_high, wHp);
    pack_w_low2<<<576,  256, 0, stream>>>(w_low2, wL2p);
    pack_w_low1<<<64,   256, 0, stream>>>(w_low1, wL1p);
    mid_kernel <<<256,  256, 0, stream>>>(inx, b_low1, wL1p, midp);
    tconv_kernel<<<NB * 64 * 4, 256, 0, stream>>>(
        inx, mask, inv_mask, b_high, b_low2, wHp, wL2p, midp, out);
}
